// StructureDecoder_22385369547415
// MI455X (gfx1250) — matrix operand order repaired and checked by reading, not compiled, not run on hardware
//
#include <hip/hip_runtime.h>
#include <hip/hip_bf16.h>
#include <stdint.h>

#define N_NODES 8192
#define D_FEAT  64

typedef __attribute__((ext_vector_type(16))) __bf16        v16bf;
typedef __attribute__((ext_vector_type(8)))  float         v8f;
typedef __attribute__((ext_vector_type(8)))  unsigned int  v8u;

// ---------------------------------------------------------------- init
__global__ void init_kernel(float* __restrict__ agg, float* __restrict__ deg) {
    int idx = blockIdx.x * blockDim.x + threadIdx.x;
    if (idx < N_NODES * D_FEAT) agg[idx] = 0.0f;
    if (idx < N_NODES)          deg[idx] = 1.0f;   // self-loop contribution
}

// ------------------------------------------------------------- degrees
__global__ void degree_kernel(const int* __restrict__ ei, int E,
                              float* __restrict__ deg) {
    int e = blockIdx.x * blockDim.x + threadIdx.x;
    if (e < E) atomicAdd(&deg[ei[E + e]], 1.0f);   // dst in-degree
}

__global__ void dinv_kernel(const float* __restrict__ deg,
                            float* __restrict__ dinv) {
    int i = blockIdx.x * blockDim.x + threadIdx.x;
    if (i < N_NODES) {
        float d = deg[i];
        dinv[i] = (d > 0.0f) ? rsqrtf(d) : 0.0f;
    }
}

// --------------------------------------------------------- h = x @ W
__global__ __launch_bounds__(256) void linear_kernel(
        const float* __restrict__ x, const float* __restrict__ W,
        float* __restrict__ h) {
    __shared__ float sW[D_FEAT * D_FEAT];
    for (int i = threadIdx.x; i < D_FEAT * D_FEAT; i += blockDim.x) sW[i] = W[i];
    __syncthreads();
    int idx = blockIdx.x * blockDim.x + threadIdx.x;   // exact grid, no bound check
    int row = idx >> 6;
    int d   = idx & 63;
    const float* xr = x + row * D_FEAT;
    float acc = 0.0f;
#pragma unroll
    for (int k = 0; k < D_FEAT; ++k) acc = fmaf(xr[k], sW[k * D_FEAT + d], acc);
    h[idx] = acc;
}

// ------------------------------------------- msg scatter (edge atomics)
__global__ void scatter_kernel(const int* __restrict__ ei, int E,
                               const float* __restrict__ h,
                               const float* __restrict__ dinv,
                               float* __restrict__ agg) {
    long long idx = (long long)blockIdx.x * blockDim.x + threadIdx.x;
    int e = (int)(idx >> 6);
    int d = (int)(idx & 63);
    if (e >= E) return;
    int s = ei[e];
    int t = ei[E + e];
    float norm = dinv[s] * dinv[t];
    atomicAdd(&agg[t * D_FEAT + d], norm * h[s * D_FEAT + d]);
}

// ---------------------- self-loop + bias + ReLU, pack to bf16 for WMMA
__global__ void relu_pack_kernel(const float* __restrict__ agg,
                                 const float* __restrict__ h,
                                 const float* __restrict__ dinv,
                                 const float* __restrict__ b,
                                 __hip_bfloat16* __restrict__ h16) {
    int idx = blockIdx.x * blockDim.x + threadIdx.x;   // exact grid
    int row = idx >> 6;
    int d   = idx & 63;
    float di = dinv[row];
    float v  = agg[idx] + di * di * h[idx] + b[d];
    v = v > 0.0f ? v : 0.0f;
    h16[idx] = __float2bfloat16(v);
}

// ---------------------------------------- Gram matrix: out = h16 @ h16^T
// One wave (32 lanes) computes one 16x16 f32 tile via two
// v_wmma_f32_16x16x32_bf16 ops (K = 64 = 2 x 32).
__global__ __launch_bounds__(256) void gram_wmma_kernel(
        const __hip_bfloat16* __restrict__ h16, float* __restrict__ out) {
    const unsigned lane = threadIdx.x & 31u;
    const unsigned wave = threadIdx.x >> 5u;
    const unsigned tm   = blockIdx.x;                 // 0..511 row tile
    const unsigned tn   = blockIdx.y * 8u + wave;     // 0..511 col tile
    const unsigned ml   = lane & 15u;
    const unsigned hi   = lane >> 4u;                 // half-wave select

    const unsigned short* hs = (const unsigned short*)h16;
    const unsigned rowbase = tm * 16u;
    const unsigned colbase = tn * 16u;

    v8f acc = {};
#pragma unroll
    for (int k0 = 0; k0 < D_FEAT; k0 += 32) {
        v8u au, bu;
#pragma unroll
        for (int v = 0; v < 8; ++v) {
            // A operand k order used here (contiguous, same as B):
            // lanes 0-15: VGPR v -> K = 2v, 2v+1
            // lanes 16-31: same +16
            unsigned ka = (unsigned)k0 + (hi << 4) + ((unsigned)v << 1);
            au[v] = *(const unsigned*)(hs + (size_t)(rowbase + ml) * D_FEAT + ka);
            // B operand k order used here (same as A): lane = column N, VGPR v -> K=2v,2v+1,
            // hi-half lanes cover K+16.  B[k][n] = h[colbase+n][k] (== A^T).
            unsigned kb = (unsigned)k0 + (hi << 4) + ((unsigned)v << 1);
            bu[v] = *(const unsigned*)(hs + (size_t)(colbase + ml) * D_FEAT + kb);
        }
        v16bf a = __builtin_bit_cast(v16bf, au);
        v16bf b = __builtin_bit_cast(v16bf, bu);
        acc = __builtin_amdgcn_wmma_f32_16x16x32_bf16(
                  false, a, false, b, (short)0, acc, false, false);
    }
    // C/D 16x16 f32 layout: VGPR j -> M = j + hi*8, N = ml
#pragma unroll
    for (int j = 0; j < 8; ++j) {
        out[(size_t)(rowbase + (unsigned)j + hi * 8u) * N_NODES + colbase + ml] = acc[j];
    }
}

// ---------------------------------------------------------------- launch
extern "C" void kernel_launch(void* const* d_in, const int* in_sizes, int n_in,
                              void* d_out, int out_size, void* d_ws, size_t ws_size,
                              hipStream_t stream) {
    const float* x  = (const float*)d_in[0];
    const int*   ei = (const int*)d_in[1];
    const float* W  = (const float*)d_in[2];
    const float* b  = (const float*)d_in[3];
    float*       out = (float*)d_out;
    const int E = in_sizes[1] / 2;

    char* ws = (char*)d_ws;
    float* deg  = (float*)(ws);
    float* dinv = (float*)(ws + 32768);
    float* h    = (float*)(ws + 65536);
    float* agg  = (float*)(ws + 65536 + (size_t)N_NODES * D_FEAT * 4);
    __hip_bfloat16* h16 =
        (__hip_bfloat16*)(ws + 65536 + 2 * (size_t)N_NODES * D_FEAT * 4);

    const int NB_ND = (N_NODES * D_FEAT) / 256;           // 2048

    init_kernel<<<NB_ND, 256, 0, stream>>>(agg, deg);
    degree_kernel<<<(E + 255) / 256, 256, 0, stream>>>(ei, E, deg);
    dinv_kernel<<<(N_NODES + 255) / 256, 256, 0, stream>>>(deg, dinv);
    linear_kernel<<<NB_ND, 256, 0, stream>>>(x, W, h);

    long long tot = (long long)E * D_FEAT;
    int sblocks = (int)((tot + 255) / 256);
    scatter_kernel<<<sblocks, 256, 0, stream>>>(ei, E, h, dinv, agg);

    relu_pack_kernel<<<NB_ND, 256, 0, stream>>>(agg, h, dinv, b, h16);

    dim3 grid(N_NODES / 16, N_NODES / 16 / 8);            // 512 x 64, 8 waves/block
    gram_wmma_kernel<<<grid, 256, 0, stream>>>(h16, out);
}
